// MultiHeadAttention_15015205667276
// MI455X (gfx1250) — matrix operand order repaired and checked by reading, not compiled, not run on hardware
//
#include <hip/hip_runtime.h>
#include <hip/hip_bf16.h>

// ---------------- CDNA5 vector types ----------------
typedef __attribute__((ext_vector_type(16))) _Float16 v16h;
typedef __attribute__((ext_vector_type(8)))  float    v8f;
typedef __attribute__((ext_vector_type(4)))  unsigned int u32x4;
typedef __attribute__((ext_vector_type(4)))  int      i32x4;
typedef __attribute__((ext_vector_type(8)))  int      i32x8;
typedef __attribute__((ext_vector_type(4)))  float    f32x4;

#define WMMA_F32_F16(a, b, c) \
  __builtin_amdgcn_wmma_f32_16x16x32_f16(false, (a), false, (b), (short)0, (c), false, false)

#if __has_builtin(__builtin_amdgcn_tensor_load_to_lds)
#define HAVE_TDM 1
#else
#define HAVE_TDM 0
#endif

// ---------------- problem constants ----------------
#define Bb   4
#define Ss   2048
#define Dd   1024
#define Hh   16
#define DHh  64
#define Mrows (Bb * Ss)                    // 8192
#define ATTN_SCALE 0.022097086912079608f   // 1/sqrt(2048): ref scales by sqrt(seq_len)

// ---------------- TDM: 2D f16 tile Global -> LDS ----------------
// Descriptor per CDNA5 ISA §8.3/§8.4: group0 = {flags, lds_addr, global_addr, type=2},
// group1 = {mask/data_size, tensor dims, tile dims, dim0 stride}. groups 2/3 zero (<=2D).
static __device__ __forceinline__ unsigned lds_addr_u32(const void* p) {
  // generic LDS pointer: low 32 bits are the LDS byte address (ISA §10.2 aperture rule)
  return (unsigned)(unsigned long long)p;
}

static __device__ __forceinline__ void tdm_load_tile_f16(
    const void* lds_dst, const void* gsrc,
    unsigned tensor_w, unsigned tensor_h, unsigned row_stride,
    unsigned tile_w, unsigned tile_h) {
#if HAVE_TDM
  unsigned long long ga = (unsigned long long)gsrc;
  u32x4 g0 = {
      1u,                                               // count=1, user mode, no gather
      lds_addr_u32(lds_dst),                            // lds_addr (bytes)
      (unsigned)(ga & 0xFFFFFFFFu),                     // global_addr[31:0]
      (unsigned)((ga >> 32) & 0x01FFFFFFu) | (2u << 30) // global_addr[56:32], type=2
  };
  i32x8 g1 = {
      (int)(1u << 16),                                        // wg_mask=0, data_size=2B
      (int)((tensor_w & 0xFFFFu) << 16),                      // tensor_dim0[15:0]
      (int)(((tensor_w >> 16) & 0xFFFFu) | ((tensor_h & 0xFFFFu) << 16)),
      (int)(((tensor_h >> 16) & 0xFFFFu) | ((tile_w & 0xFFFFu) << 16)),  // tile_dim0
      (int)(tile_h & 0xFFFFu),                                // tile_dim1, tile_dim2=0
      (int)row_stride,                                        // tensor_dim0_stride[31:0]
      0, 0                                                    // stride hi, dim1_stride
  };
  i32x4 z4 = {0, 0, 0, 0};
#if __has_include(<hip/amd_detail/amd_gfx1250_TDM.h>)
  i32x8 z8 = {0, 0, 0, 0, 0, 0, 0, 0};
  __builtin_amdgcn_tensor_load_to_lds(g0, g1, z4, z4, z8, 0); // 6-arg toolchain
#else
  __builtin_amdgcn_tensor_load_to_lds(g0, g1, z4, z4, 0);     // 5-arg toolchain
#endif
#else
  (void)lds_dst; (void)gsrc; (void)tensor_w; (void)tensor_h;
  (void)row_stride; (void)tile_w; (void)tile_h;
#endif
}

static __device__ __forceinline__ void tdm_wait() {
#if HAVE_TDM
#if __has_builtin(__builtin_amdgcn_s_wait_tensorcnt)
  __builtin_amdgcn_s_wait_tensorcnt(0);
#endif
#endif
}

// ---------------- fragment loaders ----------------
// A fragment (16x32 f16): lane (g = lane>>4, mr = lane&15) holds row m0+mr.
// v0..v3: K = g*8 + 0..7 ; v4..v7: K = 16 + g*8 + 0..7 (two 16B contiguous loads)
static __device__ __forceinline__ v16h ld_afrag(const _Float16* tile, int ld, int m0, int k0) {
  const int lane = threadIdx.x & 31;
  const int g = lane >> 4, mr = lane & 15;
  const _Float16* p = tile + (size_t)(m0 + mr) * ld + k0 + g * 8;
  union { v16h v; u32x4 q[2]; } u;
  u.q[0] = *(const u32x4*)(p);
  u.q[1] = *(const u32x4*)(p + 16);
  return u.v;
}

// B fragment (32x16 f16): lane nc holds column n0+nc; lane-half g selects
// K = g*8 + 0..7 and 16 + g*8 + 0..7 (two 16B loads when tile stored [n][k]).
static __device__ __forceinline__ v16h ld_bfrag(const _Float16* tile, int ld, int n0, int k0) {
  const int lane = threadIdx.x & 31;
  const int g = lane >> 4, nc = lane & 15;
  const _Float16* p = tile + (size_t)(n0 + nc) * ld + k0 + g * 8;
  union { v16h v; u32x4 q[2]; } u;
  u.q[0] = *(const u32x4*)(p);
  u.q[1] = *(const u32x4*)(p + 16);
  return u.v;
}

// ---------------- f32 -> f16 conversion (8 elems/thread/iter) ----------------
__global__ void __launch_bounds__(256) cvt_f32_f16(const float* __restrict__ src,
                                                   _Float16* __restrict__ dst, int n) {
  const int stride = gridDim.x * blockDim.x * 8;
  for (int i = (blockIdx.x * blockDim.x + threadIdx.x) * 8; i < n; i += stride) {
    f32x4 a = *(const f32x4*)(src + i);
    f32x4 b = *(const f32x4*)(src + i + 4);
    union { _Float16 h[8]; u32x4 q; } u;
#pragma unroll
    for (int j = 0; j < 4; ++j) { u.h[j] = (_Float16)a[j]; u.h[4 + j] = (_Float16)b[j]; }
    *(u32x4*)(dst + i) = u.q;
  }
}

// ---------------- tiled WMMA GEMM: C[M,N] = A[M,K] @ B[K,N] + bias ----------------
// 256 threads = 8 waves; block tile 128x128; wave computes 32x64 (8 WMMAs / k-step).
// A tile staged by TDM (straight 2D copy), B tile transposed in LDS by VALU/ds.
template <int STORE_HALF>
__global__ void __launch_bounds__(256) gemm_bias(const _Float16* __restrict__ A,
                                                 const _Float16* __restrict__ Bw,
                                                 const float* __restrict__ bias,
                                                 _Float16* __restrict__ Ch,
                                                 float* __restrict__ Cf,
                                                 int M, int N, int K) {
  __shared__ _Float16 As[128 * 32];   // [m][k] row-major (TDM target)
  __shared__ _Float16 Bs[128 * 32];   // [n][k] transposed for contiguous B-frags

  const int tid  = threadIdx.x;
  const int wave = tid >> 5;
  const int lane = tid & 31;
  const int m0 = blockIdx.y * 128;
  const int n0 = blockIdx.x * 128;
  const int wm = (wave >> 1) * 32;
  const int wn = (wave & 1) * 64;

  v8f acc[2][4] = {};

  for (int k0 = 0; k0 < K; k0 += 32) {
    __syncthreads();   // previous iteration's fragment reads complete
    if (HAVE_TDM) {
      if (wave == 0)
        tdm_load_tile_f16(As, &A[(size_t)m0 * K + k0], K, M, K, 32, 128);
    } else {
      // manual A stage: 512 chunks of 8 halves
#pragma unroll
      for (int i = 0; i < 2; ++i) {
        int c = tid + i * 256;
        int r = c >> 2, cc = (c & 3) * 8;
        *(u32x4*)&As[r * 32 + cc] = *(const u32x4*)&A[(size_t)(m0 + r) * K + k0 + cc];
      }
    }
    // B stage transposed: read Bw[k][n] (coalesced), write Bs[n][k]
#pragma unroll
    for (int i = 0; i < 2; ++i) {
      int c = tid + i * 256;
      int kk = c >> 4, nn = (c & 15) * 8;
      union { u32x4 q; _Float16 h[8]; } t;
      t.q = *(const u32x4*)&Bw[(size_t)(k0 + kk) * N + n0 + nn];
#pragma unroll
      for (int j = 0; j < 8; ++j) Bs[(nn + j) * 32 + kk] = t.h[j];
      if (k0 + 32 < K)   // prefetch next B tile line -> global_prefetch_b8
        __builtin_prefetch(&Bw[(size_t)(k0 + 32 + kk) * N + n0 + nn], 0, 0);
    }
    if (HAVE_TDM && wave == 0) tdm_wait();
    __syncthreads();

    v16h a0 = ld_afrag(As, 32, wm, 0);
    v16h a1 = ld_afrag(As, 32, wm + 16, 0);
#pragma unroll
    for (int ni = 0; ni < 4; ++ni) {
      v16h b = ld_bfrag(Bs, 32, wn + ni * 16, 0);
      acc[0][ni] = WMMA_F32_F16(a0, b, acc[0][ni]);
      acc[1][ni] = WMMA_F32_F16(a1, b, acc[1][ni]);
    }
  }

  // epilogue: lane nc = column, vgpr j -> row (j + 8*lane_half); stride-N walk
  const int g = lane >> 4, nc = lane & 15;
#pragma unroll
  for (int mi = 0; mi < 2; ++mi)
#pragma unroll
    for (int ni = 0; ni < 4; ++ni) {
      const int col = n0 + wn + ni * 16 + nc;
      const float bv = bias[col];
      const size_t base = (size_t)(m0 + wm + mi * 16 + g * 8) * N + col;
#pragma unroll
      for (int j = 0; j < 8; ++j) {
        float v = acc[mi][ni][j] + bv;
        if (STORE_HALF) Ch[base + (size_t)j * N] = (_Float16)v;
        else            Cf[base + (size_t)j * N] = v;
      }
    }
}

// ---------------- fused flash attention ----------------
// Block = (64 q-rows, head, batch); 128 threads = 4 waves; wave owns 16 q rows.
__global__ void __launch_bounds__(128) attn_fused(const _Float16* __restrict__ Qp,
                                                  const _Float16* __restrict__ Kp,
                                                  const _Float16* __restrict__ Vp,
                                                  _Float16* __restrict__ AO) {
  __shared__ _Float16 Ks[64 * 64];   // [t][dq]   B-frag: n=t, k=dq contiguous (TDM target)
  __shared__ _Float16 Vs[64 * 64];   // [dv][t]   B-frag: n=dv, k=t contiguous
  __shared__ _Float16 Ps[64 * 64];   // [qrow][t] A-frag rows

  const int tid  = threadIdx.x;
  const int wave = tid >> 5;
  const int lane = tid & 31;
  const int g = lane >> 4, nc = lane & 15;

  const int qb = blockIdx.x * 64;
  const int h  = blockIdx.y;
  const int b  = blockIdx.z;
  const size_t baseRow = (size_t)b * Ss;
  const int hoff = h * DHh;
  const int wrow = wave * 16;

  const _Float16* qtile = Qp + (baseRow + qb) * Dd + hoff;
  v16h qf0 = ld_afrag(qtile, Dd, wrow, 0);
  v16h qf1 = ld_afrag(qtile, Dd, wrow, 32);

  float mrow[8], lrow[8];
  v8f o[4] = {};
#pragma unroll
  for (int j = 0; j < 8; ++j) { mrow[j] = -1e30f; lrow[j] = 0.f; }

  for (int tb = 0; tb < Ss; tb += 64) {
    __syncthreads();   // all waves done reading Ks/Vs of previous block
    if (HAVE_TDM) {
      if (wave == 0)
        tdm_load_tile_f16(Ks, &Kp[(baseRow + tb) * Dd + hoff], Dd, Mrows, Dd, 64, 64);
    } else {
#pragma unroll
      for (int i = 0; i < 4; ++i) {
        int c = tid + i * 128;
        int r = c >> 3, cc = (c & 7) * 8;
        *(u32x4*)&Ks[r * 64 + cc] =
            *(const u32x4*)&Kp[(baseRow + tb + r) * Dd + hoff + cc];
      }
    }
    // V block transposed: coalesced global reads, scattered ds writes
#pragma unroll
    for (int i = 0; i < 4; ++i) {
      int c = tid + i * 128;
      int t = c >> 3, dv0 = (c & 7) * 8;
      union { u32x4 q; _Float16 hh[8]; } tv;
      tv.q = *(const u32x4*)&Vp[(baseRow + tb + t) * Dd + hoff + dv0];
#pragma unroll
      for (int j = 0; j < 8; ++j) Vs[(dv0 + j) * 64 + t] = tv.hh[j];
      if (tb + 64 < Ss) {   // prefetch next K/V blocks
        __builtin_prefetch(&Vp[(baseRow + tb + 64 + t) * Dd + hoff + dv0], 0, 0);
        __builtin_prefetch(&Kp[(baseRow + tb + 64 + t) * Dd + hoff + dv0], 0, 0);
      }
    }
    if (HAVE_TDM && wave == 0) tdm_wait();
    __syncthreads();

    // S_tile = (Q @ K^T) * scale
    v8f st[4];
#pragma unroll
    for (int nf = 0; nf < 4; ++nf) {
      v8f s = {};
      s = WMMA_F32_F16(qf0, ld_bfrag(Ks, 64, nf * 16, 0),  s);
      s = WMMA_F32_F16(qf1, ld_bfrag(Ks, 64, nf * 16, 32), s);
      st[nf] = s * ATTN_SCALE;
    }

    // row max across frags, then across the 16-lane column group
    float rmax[8];
#pragma unroll
    for (int j = 0; j < 8; ++j)
      rmax[j] = fmaxf(fmaxf(st[0][j], st[1][j]), fmaxf(st[2][j], st[3][j]));
#pragma unroll
    for (int d = 1; d < 16; d <<= 1)
#pragma unroll
      for (int j = 0; j < 8; ++j)
        rmax[j] = fmaxf(rmax[j], __shfl_xor(rmax[j], d, 32));

    float corr[8], mnew[8];
#pragma unroll
    for (int j = 0; j < 8; ++j) {
      mnew[j] = fmaxf(mrow[j], rmax[j]);
      corr[j] = __expf(mrow[j] - mnew[j]);
      mrow[j] = mnew[j];
      lrow[j] *= corr[j];
    }
#pragma unroll
    for (int f = 0; f < 4; ++f)
#pragma unroll
      for (int j = 0; j < 8; ++j) o[f][j] *= corr[j];

    // P = exp(S - m) -> f16 in LDS (wave-private rows)
    float psum[8] = {0.f, 0.f, 0.f, 0.f, 0.f, 0.f, 0.f, 0.f};
#pragma unroll
    for (int nf = 0; nf < 4; ++nf)
#pragma unroll
      for (int j = 0; j < 8; ++j) {
        float pv = __expf(st[nf][j] - mnew[j]);
        psum[j] += pv;
        Ps[(wrow + j + g * 8) * 64 + nf * 16 + nc] = (_Float16)pv;
      }
#pragma unroll
    for (int d = 1; d < 16; d <<= 1)
#pragma unroll
      for (int j = 0; j < 8; ++j) psum[j] += __shfl_xor(psum[j], d, 32);
#pragma unroll
    for (int j = 0; j < 8; ++j) lrow[j] += psum[j];

    // O += P @ V
    v16h pa0 = ld_afrag(Ps, 64, wrow, 0);
    v16h pa1 = ld_afrag(Ps, 64, wrow, 32);
#pragma unroll
    for (int df = 0; df < 4; ++df) {
      o[df] = WMMA_F32_F16(pa0, ld_bfrag(Vs, 64, df * 16, 0),  o[df]);
      o[df] = WMMA_F32_F16(pa1, ld_bfrag(Vs, 64, df * 16, 32), o[df]);
    }
  }

  // epilogue: O / l -> AO[b*S+s][h*64+dv] (f16 feeds output projection)
#pragma unroll
  for (int df = 0; df < 4; ++df) {
    const size_t base = (baseRow + qb + wrow + g * 8) * Dd + hoff + df * 16 + nc;
#pragma unroll
    for (int j = 0; j < 8; ++j)
      AO[base + (size_t)j * Dd] = (_Float16)(o[df][j] / lrow[j]);
  }
}

// ---------------- host-side orchestration ----------------
extern "C" void kernel_launch(void* const* d_in, const int* in_sizes, int n_in,
                              void* d_out, int out_size, void* d_ws, size_t ws_size,
                              hipStream_t stream) {
  (void)in_sizes; (void)n_in; (void)out_size; (void)ws_size;

  const float* query = (const float*)d_in[0];
  const float* key   = (const float*)d_in[1];
  const float* value = (const float*)d_in[2];
  const float* Wq_w  = (const float*)d_in[3];
  const float* Wq_b  = (const float*)d_in[4];
  const float* Wk_w  = (const float*)d_in[5];
  const float* Wk_b  = (const float*)d_in[6];
  const float* Wv_w  = (const float*)d_in[7];
  const float* Wv_b  = (const float*)d_in[8];
  const float* Wo_w  = (const float*)d_in[9];
  const float* Wo_b  = (const float*)d_in[10];

  const size_t MS  = (size_t)Mrows * Dd;
  const size_t WSZ = (size_t)Dd * Dd;

  _Float16* p   = (_Float16*)d_ws;
  _Float16* Xq  = p; p += MS;
  _Float16* Xk  = p; p += MS;
  _Float16* Xv  = p; p += MS;
  _Float16* Wqh = p; p += WSZ;
  _Float16* Wkh = p; p += WSZ;
  _Float16* Wvh = p; p += WSZ;
  _Float16* Woh = p; p += WSZ;
  _Float16* Qp  = p; p += MS;
  _Float16* Kp  = p; p += MS;
  _Float16* Vp  = p; p += MS;
  _Float16* AO  = p; p += MS;

  dim3 cb(256), cg(512);
  cvt_f32_f16<<<cg, cb, 0, stream>>>(query, Xq, (int)MS);
  cvt_f32_f16<<<cg, cb, 0, stream>>>(key,   Xk, (int)MS);
  cvt_f32_f16<<<cg, cb, 0, stream>>>(value, Xv, (int)MS);
  cvt_f32_f16<<<cg, cb, 0, stream>>>(Wq_w,  Wqh, (int)WSZ);
  cvt_f32_f16<<<cg, cb, 0, stream>>>(Wk_w,  Wkh, (int)WSZ);
  cvt_f32_f16<<<cg, cb, 0, stream>>>(Wv_w,  Wvh, (int)WSZ);
  cvt_f32_f16<<<cg, cb, 0, stream>>>(Wo_w,  Woh, (int)WSZ);

  dim3 gb(256), gg(Dd / 128, Mrows / 128);
  gemm_bias<1><<<gg, gb, 0, stream>>>(Xq, Wqh, Wq_b, Qp, nullptr, Mrows, Dd, Dd);
  gemm_bias<1><<<gg, gb, 0, stream>>>(Xk, Wkh, Wk_b, Kp, nullptr, Mrows, Dd, Dd);
  gemm_bias<1><<<gg, gb, 0, stream>>>(Xv, Wvh, Wv_b, Vp, nullptr, Mrows, Dd, Dd);

  dim3 ab(128), ag(Ss / 64, Hh, Bb);
  attn_fused<<<ag, ab, 0, stream>>>(Qp, Kp, Vp, AO);

  gemm_bias<0><<<gg, gb, 0, stream>>>(AO, Woh, Wo_b, nullptr, (float*)d_out, Mrows, Dd, Dd);
}
